// MMDLoss_20246475833388
// MI455X (gfx1250) — compile-verified, matrix operand order checked by reading, not run on hardware
//
#include <hip/hip_runtime.h>
#include <hip/hip_bf16.h>

// CDNA5 WMMA operand/accumulator vector types (wave32 fragments)
typedef __attribute__((ext_vector_type(16))) __bf16 v16bf;
typedef __attribute__((ext_vector_type(8)))  __bf16 v8bf;
typedef __attribute__((ext_vector_type(4)))  __bf16 v4bf;
typedef __attribute__((ext_vector_type(8)))  float  v8f;

#define NROWS 8192
#define NDIM  512

// Workspace layout (bytes):
//   [0,               65536)  : fp32 row norms (s rows 0..8191, t rows 8192..16383)
//   [65536,           65544)  : f64 accumulator
//   [65600, 65600+4*8388608)  : bf16 s_hi | s_lo | t_hi | t_lo, each [8192][512]
#define WS_NORMS_BYTES  (2 * NROWS * sizeof(float))
#define WS_BF16_OFF     (WS_NORMS_BYTES + 64)
#define WS_BF16_ONE     ((size_t)NROWS * NDIM * sizeof(__bf16))

// exp(-d/(2*bw^2)) = exp2(-d * 1/(2*bw^2*ln2)) for bw = 0.2,0.5,sqrt(0.5),2.0,5.0.
// constexpr (not __constant__) so the unrolled exp chain uses inline literals.
constexpr float kCoef[5] = {
    18.0336880f, 2.88539008f, 1.44269504f, 0.180336880f, 0.0288539008f
};

// --- fragment loaders --------------------------------------------------------
// ISA layout (16-bit A/B operand, 16x32 MxK): lane L holds row M=L%16; lanes
// 0-15 carry K chunks [k..k+7] and [k+16..k+23]; lanes 16-31 carry
// [k+8..k+15] and [k+24..k+31]. Vector elems 0..7 = chunk0, 8..15 = chunk1.

// Staged path: two 16-byte loads straight from the pre-split bf16 arrays.
__device__ __forceinline__ v16bf load_frag_bf16(const __bf16* __restrict__ row,
                                                int k0) {
    union { struct { v8bf c0; v8bf c1; } h; v16bf v; } u;
    u.h.c0 = *reinterpret_cast<const v8bf*>(row + k0);
    u.h.c1 = *reinterpret_cast<const v8bf*>(row + k0 + 16);
    return u.v;
}

// Fallback path: load fp32 and split hi/lo in registers.
__device__ __forceinline__ void load_frag_hilo(const float* __restrict__ base,
                                               int row, int k0,
                                               v16bf& hi, v16bf& lo) {
    const float* p = base + (size_t)row * NDIM;
    float4 x0 = *reinterpret_cast<const float4*>(p + k0);
    float4 x1 = *reinterpret_cast<const float4*>(p + k0 + 4);
    float4 x2 = *reinterpret_cast<const float4*>(p + k0 + 16);
    float4 x3 = *reinterpret_cast<const float4*>(p + k0 + 20);
    float x[16] = {x0.x, x0.y, x0.z, x0.w, x1.x, x1.y, x1.z, x1.w,
                   x2.x, x2.y, x2.z, x2.w, x3.x, x3.y, x3.z, x3.w};
#pragma unroll
    for (int i = 0; i < 16; ++i) {
        __bf16 h = (__bf16)x[i];
        hi[i] = h;
        lo[i] = (__bf16)(x[i] - (float)h);
    }
}

// --- pass 1: norms + accumulator zero + (optional) bf16 hi/lo staging -------
// One wave32 per row; each lane owns 4 float4 chunks of the 512-wide row.
__global__ __launch_bounds__(256)
void mmd_stage_kernel(const float* __restrict__ s, const float* __restrict__ t,
                      float* __restrict__ norms,
                      __bf16* __restrict__ shi, __bf16* __restrict__ slo,
                      __bf16* __restrict__ thi, __bf16* __restrict__ tlo,
                      double* __restrict__ accum, int doStage) {
    if (blockIdx.x == 0 && threadIdx.x == 0) *accum = 0.0;
    const int lane = threadIdx.x & 31;
    const int wave = threadIdx.x >> 5;
    const int row  = blockIdx.x * 8 + wave;                // 0..16383
    const bool isS = row < NROWS;
    const int  r   = isS ? row : row - NROWS;
    const float* p = (isS ? s : t) + (size_t)r * NDIM;
    __bf16* hp = (isS ? shi : thi) + (size_t)r * NDIM;
    __bf16* lp = (isS ? slo : tlo) + (size_t)r * NDIM;

    float acc = 0.0f;
#pragma unroll
    for (int it = 0; it < 4; ++it) {
        const int j = lane * 4 + it * 128;
        float4 v = *reinterpret_cast<const float4*>(p + j);
        acc += v.x * v.x + v.y * v.y + v.z * v.z + v.w * v.w;
        if (doStage) {
            float xs[4] = {v.x, v.y, v.z, v.w};
            v4bf h, l;
#pragma unroll
            for (int e = 0; e < 4; ++e) {
                __bf16 hb = (__bf16)xs[e];
                h[e] = hb;
                l[e] = (__bf16)(xs[e] - (float)hb);
            }
            *reinterpret_cast<v4bf*>(hp + j) = h;
            *reinterpret_cast<v4bf*>(lp + j) = l;
        }
    }
#pragma unroll
    for (int off = 16; off > 0; off >>= 1)
        acc += __shfl_xor(acc, off, 32);
    if (lane == 0) norms[row] = acc;
}

// --- pass 2: WMMA GEMM + fused multi-bandwidth Gaussian epilogue ------------
// Block = 8 waves arranged 2(M) x 4(N); each wave owns a 32x64 macro-tile
// (2x4 WMMA tiles -> 24 v_wmma per K-step, 1 fragment load per 2 v_wmma).
// grid.z: 0=(s,s,+1) 1=(t,t,+1) 2=(s,t,-2).
// For the symmetric pairs (z=0,1) only the lower triangle is computed:
// strictly-upper blocks exit immediately; elements are weighted 2/1/0 for
// n<m / n==m / n>m, which exactly reproduces the full-matrix sum.
template <int STAGED>
__global__ __launch_bounds__(256)
void mmd_tile_kernel(const float* __restrict__ s, const float* __restrict__ t,
                     const __bf16* __restrict__ shi, const __bf16* __restrict__ slo,
                     const __bf16* __restrict__ thi, const __bf16* __restrict__ tlo,
                     const float* __restrict__ norms,
                     double* __restrict__ accum) {
    const bool sym = (blockIdx.z != 2);
    const int blockM = blockIdx.y * 64;
    const int blockN = blockIdx.x * 256;
    if (sym && blockN >= blockM + 64) return;   // strictly above the diagonal

    const int lane = threadIdx.x & 31;
    const int wave = threadIdx.x >> 5;
    const int half = lane >> 4;
    const int l16  = lane & 15;

    const float *Af, *Bf, *a2p, *b2p;
    const __bf16 *Ahi, *Alo, *Bhi, *Blo;
    float w;
    if (blockIdx.z == 0) {
        Af = s; Bf = s; Ahi = shi; Alo = slo; Bhi = shi; Blo = slo;
        a2p = norms; b2p = norms; w = 1.0f;
    } else if (blockIdx.z == 1) {
        Af = t; Bf = t; Ahi = thi; Alo = tlo; Bhi = thi; Blo = tlo;
        a2p = norms + NROWS; b2p = norms + NROWS; w = 1.0f;
    } else {
        Af = s; Bf = t; Ahi = shi; Alo = slo; Bhi = thi; Blo = tlo;
        a2p = norms; b2p = norms + NROWS; w = -2.0f;
    }

    const int mBase = blockM + (wave >> 2) * 32;
    const int nBase = blockN + (wave & 3)  * 64;

    v8f c[2][4] = {};

    for (int ks = 0; ks < NDIM; ks += 32) {
        const int k0 = ks + half * 8;
        v16bf ah[2], al[2], bh[4], bl[4];
#pragma unroll
        for (int mi = 0; mi < 2; ++mi) {
            const int row = mBase + mi * 16 + l16;
            if (STAGED) {
                ah[mi] = load_frag_bf16(Ahi + (size_t)row * NDIM, k0);
                al[mi] = load_frag_bf16(Alo + (size_t)row * NDIM, k0);
            } else {
                load_frag_hilo(Af, row, k0, ah[mi], al[mi]);
            }
        }
#pragma unroll
        for (int ni = 0; ni < 4; ++ni) {
            const int row = nBase + ni * 16 + l16;
            if (STAGED) {
                bh[ni] = load_frag_bf16(Bhi + (size_t)row * NDIM, k0);
                bl[ni] = load_frag_bf16(Blo + (size_t)row * NDIM, k0);
            } else {
                load_frag_hilo(Bf, row, k0, bh[ni], bl[ni]);
            }
        }
#pragma unroll
        for (int mi = 0; mi < 2; ++mi)
#pragma unroll
            for (int ni = 0; ni < 4; ++ni) {
                // hi*hi + hi*lo + lo*hi  (~fp32-accurate dot product)
                c[mi][ni] = __builtin_amdgcn_wmma_f32_16x16x32_bf16(
                    false, ah[mi], false, bh[ni], (short)0, c[mi][ni], false, false);
                c[mi][ni] = __builtin_amdgcn_wmma_f32_16x16x32_bf16(
                    false, ah[mi], false, bl[ni], (short)0, c[mi][ni], false, false);
                c[mi][ni] = __builtin_amdgcn_wmma_f32_16x16x32_bf16(
                    false, al[mi], false, bh[ni], (short)0, c[mi][ni], false, false);
            }
    }

    // Epilogue. C/D fragment layout: VGPR r holds M = r + 8*half, N = l16.
    float acc = 0.0f;
#pragma unroll
    for (int mi = 0; mi < 2; ++mi) {
        float a2[8];
#pragma unroll
        for (int r = 0; r < 8; ++r)
            a2[r] = a2p[mBase + mi * 16 + half * 8 + r];
#pragma unroll
        for (int ni = 0; ni < 4; ++ni) {
            const int   n  = nBase + ni * 16 + l16;
            const float b2 = b2p[n];
#pragma unroll
            for (int r = 0; r < 8; ++r) {
                float d = fmaxf(a2[r] + b2 - 2.0f * c[mi][ni][r], 0.0f);
                float ksum = 0.0f;
#pragma unroll
                for (int bwi = 0; bwi < 5; ++bwi)
                    ksum += __builtin_amdgcn_exp2f(-d * kCoef[bwi]);
                float f = 1.0f;
                if (sym) {
                    const int m = mBase + mi * 16 + half * 8 + r;
                    f = (n < m) ? 2.0f : ((n == m) ? 1.0f : 0.0f);
                }
                acc = fmaf(f, ksum, acc);
            }
        }
    }

#pragma unroll
    for (int off = 16; off > 0; off >>= 1)
        acc += __shfl_xor(acc, off, 32);
    if (lane == 0)
        atomicAdd(accum, (double)(w * acc));
}

__global__ void mmd_finalize_kernel(const double* __restrict__ accum,
                                    float* __restrict__ out) {
    if (threadIdx.x == 0 && blockIdx.x == 0)
        out[0] = (float)(*accum / (5.0 * (double)NROWS * (double)NROWS));
}

extern "C" void kernel_launch(void* const* d_in, const int* in_sizes, int n_in,
                              void* d_out, int out_size, void* d_ws, size_t ws_size,
                              hipStream_t stream) {
    (void)in_sizes; (void)n_in; (void)out_size;
    const float* s = (const float*)d_in[0];
    const float* t = (const float*)d_in[1];

    float*  norms = (float*)d_ws;
    double* accum = (double*)((char*)d_ws + WS_NORMS_BYTES);
    __bf16* shi   = (__bf16*)((char*)d_ws + WS_BF16_OFF);
    __bf16* slo   = shi + (size_t)NROWS * NDIM;
    __bf16* thi   = slo + (size_t)NROWS * NDIM;
    __bf16* tlo   = thi + (size_t)NROWS * NDIM;

    const int staged = (ws_size >= WS_BF16_OFF + 4 * WS_BF16_ONE) ? 1 : 0;

    mmd_stage_kernel<<<dim3(2 * NROWS / 8), dim3(256), 0, stream>>>(
        s, t, norms, shi, slo, thi, tlo, accum, staged);

    dim3 grid(NROWS / 256, NROWS / 64, 3);   // (N-tiles, M-tiles, pair)
    if (staged)
        mmd_tile_kernel<1><<<grid, dim3(256), 0, stream>>>(
            s, t, shi, slo, thi, tlo, norms, accum);
    else
        mmd_tile_kernel<0><<<grid, dim3(256), 0, stream>>>(
            s, t, shi, slo, thi, tlo, norms, accum);

    mmd_finalize_kernel<<<1, 1, 0, stream>>>(accum, (float*)d_out);
}
